// MambaDynamicsCell_43224550867081
// MI455X (gfx1250) — matrix operand order repaired and checked by reading, not compiled, not run on hardware
//
#include <hip/hip_runtime.h>
#include <hip/hip_bf16.h>
#include <math.h>

// ---------------------------------------------------------------------------
// Mamba dynamics cell on MI455X (gfx1250), wave32 + WMMA bf16 (split hi/lo x3)
// Double-buffered LDS (dynamic, 80KB) + register-staged software pipeline.
// ---------------------------------------------------------------------------

typedef __attribute__((ext_vector_type(16))) __bf16 v16bf;
typedef __attribute__((ext_vector_type(8)))  __bf16 v8bf;
typedef __attribute__((ext_vector_type(4)))  __bf16 v4bf;
typedef __attribute__((ext_vector_type(8)))  float  v8f;

#define BATCH  4096
#define DIN    4096   // D_INNER
#define HID    2048
#define NXPROJ 160    // DT_RANK + 2*D_STATE
#define DTR    128    // DT_RANK

// Tile config: block 128x128, K-step 32, 8 waves (2 x 4), wave tile 64x32.
#define BM 128
#define BN 128
#define BKT 32
#define RS 40        // LDS row stride in bf16 elems (80B rows: 16B-aligned, conflict-free)
#define TILE_E (BM * RS)          // elems per plane (5120)
#define STAGE_E (4 * TILE_E)      // elems per stage: Ahi|Alo|Bhi|Blo (20480 bf16 = 40960 B)
#define SMEM_BYTES (2 * STAGE_E * 2)  // two stages = 81920 B

__device__ __forceinline__ v8f zero8() {
  v8f z;
#pragma unroll
  for (int i = 0; i < 8; ++i) z[i] = 0.0f;
  return z;
}

__device__ __forceinline__ v16bf ldfrag(const __bf16* p0, const __bf16* p1) {
  v8bf a = *(const v8bf*)p0;
  v8bf b = *(const v8bf*)p1;
  v16bf r;
#pragma unroll
  for (int i = 0; i < 8; ++i) { r[i] = a[i]; r[i + 8] = b[i]; }
  return r;
}

// Convert 4 fp32 values to bf16 hi + bf16 residual (lo), store 8B each to LDS.
__device__ __forceinline__ void cvt_store(float4 f, __bf16* hi_p, __bf16* lo_p) {
  float fx[4] = {f.x, f.y, f.z, f.w};
  v4bf hv, lv;
#pragma unroll
  for (int i = 0; i < 4; ++i) {
    __bf16 hb = (__bf16)fx[i];
    hv[i] = hb;
    lv[i] = (__bf16)(fx[i] - (float)hb);
  }
  *(v4bf*)hi_p = hv;
  *(v4bf*)lo_p = lv;
}

// MODE 0: OUT = silu(acc)
// MODE 1: OUT = acc            (guard n < Nmax)
// MODE 2: dt/ssm/y fused epilogue (e0=b_dt[n], e1=x_br, e2=bc[m], e3=D[n], e4=z_silu; OUT=y)
// MODE 3: OUT = acc + e0[m,n]  (h residual)
template <int MODE>
__global__ __launch_bounds__(256) void gemm_bf16x3(
    const float* A, int lda, const float* A2, int lda2,
    const float* W, int ldw, const float* W2, int ldw2,
    int ksplit, int Ktot, int Nmax,
    float* OUT, int ldo,
    const float* e0, const float* e1, const float* e2, const float* e3,
    const float* e4) {
  extern __shared__ __align__(16) char smem_raw[];
  __bf16* smem = (__bf16*)smem_raw;

  const int t    = threadIdx.x;
  const int lane = t & 31;
  const int wave = t >> 5;       // 0..7
  const int wm   = wave >> 2;    // 0..1  (64-row strip)
  const int wn   = wave & 3;     // 0..3  (32-col strip)
  const int lh   = lane >> 4;    // lane half
  const int ln   = lane & 15;

  const int c4 = t & 7;    // which float4 within the 32-wide K slab
  const int r0 = t >> 3;   // 0..31

  auto ldA = [&](int kk, int r) -> float4 {
    const float* sa; int col, ld;
    if (kk < ksplit) { sa = A;  col = kk;          ld = lda;  }
    else             { sa = A2; col = kk - ksplit; ld = lda2; }
    const int m = blockIdx.y * BM + r;
    return *(const float4*)(sa + (size_t)m * ld + col);
  };
  auto ldB = [&](int kk, int r) -> float4 {
    const float* sw; int wcol, wld;
    if (kk < ksplit) { sw = W;  wcol = kk;          wld = ldw;  }
    else             { sw = W2; wcol = kk - ksplit; wld = ldw2; }
    const int n = blockIdx.x * BN + r;
    if (n < Nmax) return *(const float4*)(sw + (size_t)n * wld + wcol);
    float4 z; z.x = 0.f; z.y = 0.f; z.z = 0.f; z.w = 0.f; return z;
  };

  v8f acc[4][2];
#pragma unroll
  for (int mt = 0; mt < 4; ++mt)
#pragma unroll
    for (int nt = 0; nt < 2; ++nt) acc[mt][nt] = zero8();

  const int KT = Ktot / BKT;

  // ---- pipeline prologue: fetch tile 0 into registers ----
  float4 fA[4], fB[4];
#pragma unroll
  for (int q = 0; q < 4; ++q) {
    const int kk = c4 * 4;
    fA[q] = ldA(kk, r0 + q * 32);
    fB[q] = ldB(kk, r0 + q * 32);
  }

  for (int kt = 0; kt < KT; ++kt) {
    __bf16* Ahi = smem + (kt & 1) * STAGE_E;
    __bf16* Alo = Ahi + TILE_E;
    __bf16* Bhi = Alo + TILE_E;
    __bf16* Blo = Bhi + TILE_E;

    // ---- stage k: convert registers -> LDS (buf k&1) ----
#pragma unroll
    for (int q = 0; q < 4; ++q) {
      const int r = r0 + q * 32;
      cvt_store(fA[q], &Ahi[r * RS + c4 * 4], &Alo[r * RS + c4 * 4]);
      cvt_store(fB[q], &Bhi[r * RS + c4 * 4], &Blo[r * RS + c4 * 4]);
    }

    // ---- issue global loads for stage k+1 (latency hidden under compute) ----
    if (kt + 1 < KT) {
      const int kk = (kt + 1) * BKT + c4 * 4;
#pragma unroll
      for (int q = 0; q < 4; ++q) {
        fA[q] = ldA(kk, r0 + q * 32);
        fB[q] = ldB(kk, r0 + q * 32);
      }
    }
    // ---- L2 prefetch for stage k+2 (one lane per 128B row segment) ----
    if (kt + 2 < KT && c4 == 0) {
      const int kk = (kt + 2) * BKT;
#pragma unroll
      for (int q = 0; q < 4; ++q) {
        const int r = r0 + q * 32;
        {
          const float* sa; int col, ld;
          if (kk < ksplit) { sa = A;  col = kk;          ld = lda;  }
          else             { sa = A2; col = kk - ksplit; ld = lda2; }
          __builtin_prefetch(sa + (size_t)(blockIdx.y * BM + r) * ld + col, 0, 1);
        }
        {
          const float* sw; int wcol, wld;
          if (kk < ksplit) { sw = W;  wcol = kk;          wld = ldw;  }
          else             { sw = W2; wcol = kk - ksplit; wld = ldw2; }
          const int n = blockIdx.x * BN + r;
          if (n < Nmax) __builtin_prefetch(sw + (size_t)n * wld + wcol, 0, 1);
        }
      }
    }

    __syncthreads();  // stage-k stores visible; buf k&1 free of stage k-2 readers

    // ---- compute: 4 M-tiles x 2 N-tiles, 3 WMMAs each (hi*hi + hi*lo + lo*hi)
    v16bf bh[2], bl[2];
#pragma unroll
    for (int nt = 0; nt < 2; ++nt) {
      const int n = wn * 32 + nt * 16 + ln;
      const __bf16* pb = &Bhi[n * RS + lh * 8];  // B frag: K = lh*8 + j (+16)
      bh[nt] = ldfrag(pb, pb + 16);
      const __bf16* pl = &Blo[n * RS + lh * 8];
      bl[nt] = ldfrag(pl, pl + 16);
    }
#pragma unroll
    for (int mt = 0; mt < 4; ++mt) {
      const int m = wm * 64 + mt * 16 + ln;
      const __bf16* pa = &Ahi[m * RS + lh * 8];   // A frag: K = lh*8 + j (+16)
      v16bf ah = ldfrag(pa, pa + 16);
      const __bf16* pal = &Alo[m * RS + lh * 8];
      v16bf al = ldfrag(pal, pal + 16);
#pragma unroll
      for (int nt = 0; nt < 2; ++nt) {
        acc[mt][nt] = __builtin_amdgcn_wmma_f32_16x16x32_bf16(
            false, ah, false, bh[nt], (short)0, acc[mt][nt], false, false);
        acc[mt][nt] = __builtin_amdgcn_wmma_f32_16x16x32_bf16(
            false, ah, false, bl[nt], (short)0, acc[mt][nt], false, false);
        acc[mt][nt] = __builtin_amdgcn_wmma_f32_16x16x32_bf16(
            false, al, false, bh[nt], (short)0, acc[mt][nt], false, false);
      }
    }
  }

  // ---- epilogue ----
#pragma unroll
  for (int mt = 0; mt < 4; ++mt) {
#pragma unroll
    for (int nt = 0; nt < 2; ++nt) {
      v8f c = acc[mt][nt];
      const int n = blockIdx.x * BN + wn * 32 + nt * 16 + ln;
#pragma unroll
      for (int i = 0; i < 8; ++i) {
        const int m = blockIdx.y * BM + wm * 64 + mt * 16 + lh * 8 + i;
        const float v = c[i];
        if constexpr (MODE == 0) {
          OUT[(size_t)m * ldo + n] = v * (1.0f / (1.0f + __expf(-v)));
        } else if constexpr (MODE == 1) {
          if (n < Nmax) OUT[(size_t)m * ldo + n] = v;
        } else if constexpr (MODE == 2) {
          const float pre = v + e0[n];
          const float dt  = (pre > 20.0f) ? pre : log1pf(__expf(pre));
          const float xb  = e1[(size_t)m * DIN + n];
          const float val = (dt * e2[m] + e3[n]) * xb;
          OUT[(size_t)m * ldo + n] = val * e4[(size_t)m * DIN + n];
        } else {
          OUT[(size_t)m * ldo + n] = v + e0[(size_t)m * ldo + n];
        }
      }
    }
  }
}

// bc[m] = sum_j B_mat[m,j] * C_mat[m,j], j in [0,16)
__global__ __launch_bounds__(256) void bc_kernel(const float* xdbl, float* bc) {
  const int m = blockIdx.x * blockDim.x + threadIdx.x;
  if (m < BATCH) {
    const float* p = xdbl + (size_t)m * NXPROJ;
    float s = 0.f;
#pragma unroll
    for (int j = 0; j < 16; ++j) s += p[DTR + j] * p[DTR + 16 + j];
    bc[m] = s;
  }
}

__global__ __launch_bounds__(256) void layernorm_k(const float* X, const float* g,
                                                   const float* b, float* out) {
  __shared__ float red[256];
  const int row = blockIdx.x;
  const int t = threadIdx.x;
  const float* x = X + (size_t)row * HID;
  float v[8];
  float s = 0.f;
#pragma unroll
  for (int i = 0; i < 8; ++i) { v[i] = x[t + i * 256]; s += v[i]; }
  red[t] = s; __syncthreads();
  for (int off = 128; off > 0; off >>= 1) {
    if (t < off) red[t] += red[t + off];
    __syncthreads();
  }
  const float mu = red[0] * (1.0f / HID);
  __syncthreads();
  float q = 0.f;
#pragma unroll
  for (int i = 0; i < 8; ++i) { const float d = v[i] - mu; q += d * d; }
  red[t] = q; __syncthreads();
  for (int off = 128; off > 0; off >>= 1) {
    if (t < off) red[t] += red[t + off];
    __syncthreads();
  }
  const float inv = rsqrtf(red[0] * (1.0f / HID) + 1e-5f);
#pragma unroll
  for (int i = 0; i < 8; ++i) {
    const int cidx = t + i * 256;
    out[(size_t)row * HID + cidx] = (v[i] - mu) * inv * g[cidx] + b[cidx];
  }
}

extern "C" void kernel_launch(void* const* d_in, const int* in_sizes, int n_in,
                              void* d_out, int out_size, void* d_ws, size_t ws_size,
                              hipStream_t stream) {
  (void)in_sizes; (void)n_in; (void)out_size; (void)ws_size;
  const float* x       = (const float*)d_in[0];
  const float* h       = (const float*)d_in[1];
  const float* W_in    = (const float*)d_in[2];
  const float* W_state = (const float*)d_in[3];
  const float* W_xproj = (const float*)d_in[4];
  const float* W_dt    = (const float*)d_in[5];
  const float* b_dt    = (const float*)d_in[6];
  /* d_in[7] = A_log: unused by the reference computation */
  const float* Dv      = (const float*)d_in[8];
  const float* W_out   = (const float*)d_in[9];
  const float* gamma   = (const float*)d_in[10];
  const float* beta    = (const float*)d_in[11];

  char* ws = (char*)d_ws;
  const size_t SZ_BIG = (size_t)BATCH * DIN * sizeof(float);  // 64 MB
  float* x_br  = (float*)(ws);                                // [B, DIN]
  float* z_y   = (float*)(ws + SZ_BIG);                       // z_silu, then y in place
  float* x_dbl = (float*)(ws + 2 * SZ_BIG);                   // [B, 160]
  float* bc    = (float*)(ws + 2 * SZ_BIG + (size_t)BATCH * NXPROJ * sizeof(float));
  float* h_new = x_br;  // x_br dead after G4; reuse for h_new

  const dim3 blk(256);
  const int BIG = 1 << 30;
  const size_t smem = SMEM_BYTES;

  // G2: z_silu = silu(x @ W_in[4096:,:].T)   (M=4096, N=4096, K=1024)
  gemm_bf16x3<0><<<dim3(DIN / BN, BATCH / BM), blk, smem, stream>>>(
      x, 1024, nullptr, 0, W_in + (size_t)DIN * 1024, 1024, nullptr, 0,
      BIG, 1024, DIN, z_y, DIN, nullptr, nullptr, nullptr, nullptr, nullptr);

  // G1: x_br = silu([x|h] @ [W_in[:4096]|W_state].T)  (K=3072, split at 1024)
  gemm_bf16x3<0><<<dim3(DIN / BN, BATCH / BM), blk, smem, stream>>>(
      x, 1024, h, 2048, W_in, 1024, W_state, 2048,
      1024, 3072, DIN, x_br, DIN, nullptr, nullptr, nullptr, nullptr, nullptr);

  // G3: x_dbl = x_br @ W_xproj.T  (N=160 with bounds, K=4096)
  gemm_bf16x3<1><<<dim3((NXPROJ + BN - 1) / BN, BATCH / BM), blk, smem, stream>>>(
      x_br, DIN, nullptr, 0, W_xproj, DIN, nullptr, 0,
      BIG, DIN, NXPROJ, x_dbl, NXPROJ, nullptr, nullptr, nullptr, nullptr, nullptr);

  // bc[m] = sum(B_mat * C_mat)
  bc_kernel<<<dim3(BATCH / 256), blk, 0, stream>>>(x_dbl, bc);

  // G4: dt GEMM (K=128) + fused softplus/ssm/silu(z) epilogue -> y (in place over z_silu)
  gemm_bf16x3<2><<<dim3(DIN / BN, BATCH / BM), blk, smem, stream>>>(
      x_dbl, NXPROJ, nullptr, 0, W_dt, DTR, nullptr, 0,
      BIG, DTR, DIN, z_y, DIN, b_dt, x_br, bc, Dv, z_y);

  // G5: h_new = y @ W_out.T + h  (M=4096, N=2048, K=4096)
  gemm_bf16x3<3><<<dim3(HID / BN, BATCH / BM), blk, smem, stream>>>(
      z_y, DIN, nullptr, 0, W_out, DIN, nullptr, 0,
      BIG, DIN, HID, h_new, HID, h, nullptr, nullptr, nullptr, nullptr);

  // LayerNorm -> d_out
  layernorm_k<<<dim3(BATCH), blk, 0, stream>>>(h_new, gamma, beta, (float*)d_out);
}
